// ModernNCA_13907104104661
// MI455X (gfx1250) — matrix operand order repaired and checked by reading, not compiled, not run on hardware
//
#include <hip/hip_runtime.h>

// ---------------- types & helpers ----------------
typedef __bf16 v16bf __attribute__((ext_vector_type(16)));
typedef float  v8f   __attribute__((ext_vector_type(8)));

#define DN 16      // numeric features
#define DC 16      // categorical features
#define KOUT 10    // classes
#define DIM 512
#define NF 32      // n_freq
#define DE 48      // d_emb
#define NCAND 40000
#define NPAD 40016
#define BQ 1024
#define KTOT 784   // 16*48 + 16
#define KPAD 800   // padded K for WMMA (25 * 32)
#define NCHUNK 8

struct __align__(16) U4 { unsigned a, b, c, d; };
struct __align__(16) F4 { float a, b, c, d; };
struct __align__(16) I4 { int a, b, c, d; };

union Frag { v16bf v; U4 q[2]; };

static __device__ __forceinline__ unsigned short f2bfu(float f) {
  unsigned u = __builtin_bit_cast(unsigned, f);
  unsigned r = u + 0x7FFFu + ((u >> 16) & 1u);   // round-to-nearest-even
  return (unsigned short)(r >> 16);
}
static __device__ __forceinline__ float bf2f(unsigned short s) {
  unsigned u = ((unsigned)s) << 16;
  return __builtin_bit_cast(float, u);
}

// A-fragment (16x32 bf16, M=16 rows). rowbase = this lane's row (m = lane&15),
// stored k-contiguous. Lane halves hold K runs [8r..8r+7] and [16+8r..16+8r+7].
static __device__ __forceinline__ v16bf ldA(const unsigned short* rowbase, int k0, int r) {
  Frag f;
  f.q[0] = *(const U4*)(rowbase + k0 + r * 8);
  f.q[1] = *(const U4*)(rowbase + k0 + 16 + r * 8);
  return f.v;
}
// B-fragment (32x16 bf16, N=16 cols). colbase = this lane's column (n = lane&15),
// stored k-contiguous. Lane halves hold K runs [8r..8r+7] and [16+8r..16+8r+7].
static __device__ __forceinline__ v16bf ldB(const unsigned short* colbase, int k0, int r) {
  Frag f;
  f.q[0] = *(const U4*)(colbase + k0 + r * 8);
  f.q[1] = *(const U4*)(colbase + k0 + 16 + r * 8);
  return f.v;
}
static __device__ __forceinline__ v8f wmma_bf16(v16bf a, v16bf b, v8f c) {
  return __builtin_amdgcn_wmma_f32_16x16x32_bf16(false, a, false, b, (short)0, c, false, false);
}

// ---------------- prep kernels ----------------
// Wt[n][k] = enc_w[k][n] as bf16, k in [0,800) zero-padded past 784.
__global__ void nca_prep_w(const float* __restrict__ enc_w, unsigned short* __restrict__ Wt) {
  int idx = blockIdx.x * blockDim.x + threadIdx.x;
  if (idx >= DIM * KPAD) return;
  int n = idx / KPAD, k = idx - n * KPAD;
  Wt[idx] = (k < KTOT) ? f2bfu(enc_w[(size_t)k * DIM + n]) : (unsigned short)0;
}
__global__ void nca_prep_y(const int* __restrict__ y, int* __restrict__ ypad) {
  int idx = blockIdx.x * blockDim.x + threadIdx.x;
  if (idx >= NPAD) return;
  int v = (idx < NCAND) ? y[idx] : 0;
  v = v < 0 ? 0 : (v > KOUT - 1 ? KOUT - 1 : v);
  ypad[idx] = v;
}
__global__ void nca_prep_cpad(unsigned* __restrict__ tail) {  // zero Cb rows [40000,40016)
  int idx = blockIdx.x * blockDim.x + threadIdx.x;
  if (idx < 16 * DIM / 2) tail[idx] = 0u;
}

// ---------------- PLR encode (16 rows / block, 4 waves) ----------------
__global__ __launch_bounds__(128) void nca_encode(
    const float* __restrict__ x_num, const float* __restrict__ x_cat,
    const float* __restrict__ freqs, const float* __restrict__ lin_w,
    const float* __restrict__ lin_b, const unsigned short* __restrict__ Wt,
    const float* __restrict__ enc_b, unsigned short* __restrict__ Hout) {
  __shared__ unsigned short sF[256][2 * NF];   // feats: 256 (row,f) pairs x 64  (32 KB)
  __shared__ unsigned short sA[16][KPAD];      // assembled A tile               (25.6 KB)
  __shared__ unsigned short sLW[DE][2 * NF];   // lin_w transposed bf16          (6 KB)
  const int row0 = blockIdx.x * 16;
  const int tid = threadIdx.x;
  const int lane = tid & 31, wid = tid >> 5;
  const int m = lane & 15, r = lane >> 4;

  // stage lin_w^T as bf16
  for (int i = tid; i < DE * 2 * NF; i += 128) {
    int j = i >> 6, k = i & 63;
    sLW[j][k] = f2bfu(lin_w[(size_t)k * DE + j]);
  }
  // trig features: pair p = row*16 + f
  for (int pp = tid; pp < 256; pp += 128) {
    int row = pp >> 4, f = pp & 15;
    float x = x_num[(size_t)(row0 + row) * DN + f];
    for (int k = 0; k < NF; ++k) {
      float t = 6.283185307179586f * freqs[f * NF + k] * x;
      sF[pp][k]      = f2bfu(__cosf(t));
      sF[pp][NF + k] = f2bfu(__sinf(t));
    }
  }
  // categorical tail + zero pad of A
  for (int i = tid; i < 16 * 32; i += 128) {
    int row = i >> 5, c = i & 31;
    sA[row][768 + c] = (c < DC) ? f2bfu(x_cat[(size_t)(row0 + row) * DC + c])
                                : (unsigned short)0;
  }
  __syncthreads();

  // emb GEMM: 16 pair-tiles (= rows) x 3 j-tiles, K=64 (2 WMMA steps), relu -> sA
  for (int t = wid; t < 48; t += 4) {
    int pt = t / 3, jt = t - pt * 3;
    v8f acc = {0.f, 0.f, 0.f, 0.f, 0.f, 0.f, 0.f, 0.f};
#pragma unroll
    for (int ks = 0; ks < 2; ++ks) {
      v16bf a = ldA(&sF[pt * 16 + m][0], ks * 32, r);
      v16bf b = ldB(&sLW[jt * 16 + m][0], ks * 32, r);
      acc = wmma_bf16(a, b, acc);
    }
    int j = jt * 16 + m;           // D-tile: lane col = emb index j
    float bj = lin_b[j];
#pragma unroll
    for (int v = 0; v < 8; ++v) {  // D-tile rows = feature index f
      int f = v + r * 8;
      float e = acc[v] + bj;
      e = e > 0.f ? e : 0.f;
      sA[pt][f * DE + j] = f2bfu(e);
    }
  }
  __syncthreads();

  // main GEMM: [16,800] @ Wt^T -> [16,512]; 8 n-tiles per wave, 25 K-steps
  for (int nt = wid * 8; nt < wid * 8 + 8; ++nt) {
    int n = nt * 16 + m;
    v8f acc = {0.f, 0.f, 0.f, 0.f, 0.f, 0.f, 0.f, 0.f};
    const unsigned short* wrow = Wt + (size_t)n * KPAD;
#pragma unroll 5
    for (int ks = 0; ks < 25; ++ks) {
      v16bf a = ldA(&sA[m][0], ks * 32, r);
      v16bf b = ldB(wrow, ks * 32, r);
      acc = wmma_bf16(a, b, acc);
    }
    float bn = enc_b[n];
#pragma unroll
    for (int v = 0; v < 8; ++v) {
      int row = v + r * 8;
      Hout[(size_t)(row0 + row) * DIM + n] = f2bfu(acc[v] + bn);
    }
  }
}

// ---------------- row norms ----------------
__global__ void nca_norms(const unsigned short* __restrict__ H, float* __restrict__ nrm, int rows) {
  int i = blockIdx.x * blockDim.x + threadIdx.x;
  if (i >= rows) return;
  const unsigned short* hp = H + (size_t)i * DIM;
  float s = 0.f;
  for (int k = 0; k < DIM; ++k) { float v = bf2f(hp[k]); s = fmaf(v, v, s); }
  nrm[i] = s;
}

// ---------------- distance + per-chunk reductions ----------------
// grid (32 b-tiles of 32 queries, 8 chunks), 4 waves; wave w owns candidate
// cols [5000*chunk + 1280*w, +len). Two 16-query WMMA B-tiles share each
// candidate A-fragment -> halves L2 traffic per FLOP.
__global__ __launch_bounds__(128) void nca_dist(
    const unsigned short* __restrict__ Qb, const unsigned short* __restrict__ Cb,
    const float* __restrict__ qn, const float* __restrict__ cn,
    const int* __restrict__ ypad, float* __restrict__ plog, float* __restrict__ psum) {
  __shared__ unsigned short sQ[32][DIM];  // 32 KB query tile (32 queries)
  __shared__ float slog[32][KOUT];
  __shared__ float ssum[32];
  const int b0 = blockIdx.x * 32;
  const int chunk = blockIdx.y;
  const int tid = threadIdx.x;
  const int lane = tid & 31, wid = tid >> 5;
  const int qi = lane & 15, r = lane >> 4;

  // stage Q tile via CDNA5 async global->LDS copy (ASYNCcnt-tracked)
  {
    unsigned ldsbase = (unsigned)(size_t)(&sQ[0][0]);  // low 32 bits = LDS offset
    const char* g = (const char*)(Qb + (size_t)b0 * DIM);
    for (int i = tid; i < 2048; i += 128) {            // 2048 x 16 B = 32 KB
      unsigned loff = ldsbase + i * 16;
      unsigned long long ga = (unsigned long long)(size_t)(g + (size_t)i * 16);
      asm volatile("global_load_async_to_lds_b128 %0, %1, off"
                   :: "v"(loff), "v"(ga) : "memory");
    }
    asm volatile("s_wait_asynccnt 0" ::: "memory");
  }
  for (int i = tid; i < 32 * KOUT; i += 128) (&slog[0][0])[i] = 0.f;
  for (int i = tid; i < 32; i += 128) ssum[i] = 0.f;
  __syncthreads();

  const float qn_l0 = qn[b0 + qi];
  const float qn_l1 = qn[b0 + 16 + qi];
  const int start = chunk * 5000 + wid * 1280;
  const int len = (wid < 3) ? 1280 : 1160;
  const int nend = start + len;
  const int ntiles = (wid < 3) ? 80 : 73;
  float sacc0 = 0.f, sacc1 = 0.f;

  for (int t = 0; t < ntiles; ++t) {
    const int tn = start + t * 16;
    const unsigned short* arow = Cb + (size_t)(tn + qi) * DIM;  // A: candidate rows
    v8f acc0 = {0.f, 0.f, 0.f, 0.f, 0.f, 0.f, 0.f, 0.f};
    v8f acc1 = {0.f, 0.f, 0.f, 0.f, 0.f, 0.f, 0.f, 0.f};
#pragma unroll 4
    for (int ks = 0; ks < 16; ++ks) {
      v16bf a  = ldA(arow, ks * 32, r);             // candidates from global (L2)
      v16bf bq0 = ldB(&sQ[qi][0], ks * 32, r);      // Q^T tile 0 from LDS
      v16bf bq1 = ldB(&sQ[16 + qi][0], ks * 32, r); // Q^T tile 1 from LDS
      acc0 = wmma_bf16(a, bq0, acc0);               // S^T: D[cand][query]
      acc1 = wmma_bf16(a, bq1, acc1);
    }
    const int nb = tn + r * 8;                       // 16B-aligned (start % 4 == 0)
    F4 c0 = *(const F4*)(cn + nb), c1 = *(const F4*)(cn + nb + 4);
    I4 y0 = *(const I4*)(ypad + nb), y1 = *(const I4*)(ypad + nb + 4);
    float cn8[8] = {c0.a, c0.b, c0.c, c0.d, c1.a, c1.b, c1.c, c1.d};
    int   ya8[8] = {y0.a, y0.b, y0.c, y0.d, y1.a, y1.b, y1.c, y1.d};
#pragma unroll
    for (int v = 0; v < 8; ++v) {
      const int n = nb + v;
      float d20 = fmaf(-2.f, acc0[v], qn_l0 + cn8[v]);
      float d21 = fmaf(-2.f, acc1[v], qn_l1 + cn8[v]);
      d20 = fmaxf(d20, 1e-12f);
      d21 = fmaxf(d21, 1e-12f);
      const float e0 = __expf(-sqrtf(d20));          // exp(-dist), TEMP = 1
      const float e1 = __expf(-sqrtf(d21));
      if (n < nend) {
        sacc0 += __expf(e0);                         // e <= 1: no max shift needed
        sacc1 += __expf(e1);
        atomicAdd(&slog[qi][ya8[v]], e0);            // class accumulation
        atomicAdd(&slog[16 + qi][ya8[v]], e1);
      }
    }
  }
  atomicAdd(&ssum[qi], sacc0);
  atomicAdd(&ssum[16 + qi], sacc1);
  __syncthreads();
  for (int i = tid; i < 32 * KOUT; i += 128) {
    int bi = i / KOUT, k = i - bi * KOUT;
    plog[((size_t)chunk * BQ + b0 + bi) * KOUT + k] = slog[bi][k];
  }
  for (int i = tid; i < 32; i += 128)
    psum[(size_t)chunk * BQ + b0 + i] = ssum[i];
}

// ---------------- finalize: out = log(sum_c plog) - sum_c log(psum) ----------------
__global__ void nca_finalize(const float* __restrict__ plog, const float* __restrict__ psum,
                             float* __restrict__ out) {
  int idx = blockIdx.x * blockDim.x + threadIdx.x;
  if (idx >= BQ * KOUT) return;
  int b = idx / KOUT, k = idx - b * KOUT;
  float lse = 0.f, lg = 0.f;
#pragma unroll
  for (int c = 0; c < NCHUNK; ++c) {
    lse += __logf(psum[(size_t)c * BQ + b]);
    lg  += plog[((size_t)c * BQ + b) * KOUT + k];
  }
  out[idx] = __logf(lg) - lse;
}

// ---------------- launch ----------------
extern "C" void kernel_launch(void* const* d_in, const int* in_sizes, int n_in,
                              void* d_out, int out_size, void* d_ws, size_t ws_size,
                              hipStream_t stream) {
  const float* x_num  = (const float*)d_in[0];
  const float* x_cat  = (const float*)d_in[1];
  const float* cx_num = (const float*)d_in[3];
  const float* cx_cat = (const float*)d_in[4];
  const int*   cy     = (const int*)d_in[5];
  const float* freqs  = (const float*)d_in[6];
  const float* lin_w  = (const float*)d_in[7];
  const float* lin_b  = (const float*)d_in[8];
  const float* enc_w  = (const float*)d_in[9];
  const float* enc_b  = (const float*)d_in[10];
  float* out = (float*)d_out;

  char* base = (char*)d_ws;
  size_t off = 0;
  auto alloc = [&](size_t bytes) -> void* {
    void* p = base + off;
    off += (bytes + 255) & ~(size_t)255;
    return p;
  };
  unsigned short* Wt = (unsigned short*)alloc((size_t)DIM * KPAD * 2);   // 0.8 MB
  unsigned short* Qb = (unsigned short*)alloc((size_t)BQ * DIM * 2);     // 1 MB
  unsigned short* Cb = (unsigned short*)alloc((size_t)NPAD * DIM * 2);   // 41 MB
  float* qn   = (float*)alloc((size_t)BQ * 4);
  float* cn   = (float*)alloc((size_t)NPAD * 4);
  int*   ypad = (int*)alloc((size_t)NPAD * 4);
  float* plog = (float*)alloc((size_t)NCHUNK * BQ * KOUT * 4);
  float* psum = (float*)alloc((size_t)NCHUNK * BQ * 4);

  nca_prep_w<<<(DIM * KPAD + 255) / 256, 256, 0, stream>>>(enc_w, Wt);
  nca_prep_y<<<(NPAD + 255) / 256, 256, 0, stream>>>(cy, ypad);
  nca_prep_cpad<<<(16 * DIM / 2 + 255) / 256, 256, 0, stream>>>(
      (unsigned*)(Cb + (size_t)NCAND * DIM));

  nca_encode<<<BQ / 16, 128, 0, stream>>>(x_num, x_cat, freqs, lin_w, lin_b, Wt, enc_b, Qb);
  nca_encode<<<NCAND / 16, 128, 0, stream>>>(cx_num, cx_cat, freqs, lin_w, lin_b, Wt, enc_b, Cb);

  nca_norms<<<(BQ + 255) / 256, 256, 0, stream>>>(Qb, qn, BQ);
  nca_norms<<<(NCAND + 255) / 256, 256, 0, stream>>>(Cb, cn, NCAND);

  nca_dist<<<dim3(BQ / 32, NCHUNK), 128, 0, stream>>>(Qb, Cb, qn, cn, ypad, plog, psum);
  nca_finalize<<<(BQ * KOUT + 255) / 256, 256, 0, stream>>>(plog, psum, out);
}
